// MultilayerLSTMScan_44040594653382
// MI455X (gfx1250) — matrix operand order repaired and checked by reading, not compiled, not run on hardware
//
#include <hip/hip_runtime.h>

typedef __attribute__((ext_vector_type(16))) __bf16 v16bf;
typedef __attribute__((ext_vector_type(8)))  __bf16 v8bf;
typedef __attribute__((ext_vector_type(8)))  float  v8f;

#define BATCH 128
#define TT    1024
#define NL    2
#define DD    512
#define HH    512
#define G4H   2048
#define KCH   16                 // 512 / 32 K-chunks per GEMM
#define LDSS  520                // padded LDS row stride (bf16 elems)

// Pre-swizzled weight layout per layer: [which(Wi=0,Wh=1)][kc][gate][colTile][512]
// Within the 512-elem fragment tile: lane-major, 16 contiguous bf16 per lane,
// in the contiguous-K order the A loads also use (lane = 16*(K>=16 half) + col,
// element e = K within half).
#define WT_PER_GEMM  (KCH * 4 * 32 * 512)   // 1,048,576 elems
#define WT_PER_LAYER (2 * WT_PER_GEMM)      // 2,097,152 elems
#define WT_TOTAL     (NL * WT_PER_LAYER)    // 4,194,304 elems

__global__ __launch_bounds__(256) void wt_prep_kernel(const float* __restrict__ Wi,
                                                      const float* __restrict__ Wh,
                                                      __bf16* __restrict__ wt) {
    int idx = blockIdx.x * 256 + threadIdx.x;
    if (idx >= WT_TOTAL) return;
    int e       = idx & 15;
    int lane    = (idx >> 4) & 31;
    int colTile = (idx >> 9) & 31;
    int g       = (idx >> 14) & 3;
    int kc      = (idx >> 16) & 15;
    int which   = (idx >> 20) & 1;
    int l       = (idx >> 21) & 1;
    int k   = kc * 32 + (lane >> 4) * 16 + e;         // K index within GEMM
    int col = g * HH + colTile * 16 + (lane & 15);    // column in [0,4H)
    const float* src = which ? Wh : Wi;               // both [L, 512, 2048]
    float v = src[((size_t)l * DD + k) * G4H + col];
    wt[idx] = (__bf16)v;
}

__device__ __forceinline__ float sigmoidf_(float x) {
    return 1.0f / (1.0f + __expf(-x));
}

// One LSTM layer, one time step. Each WG: 16 batch rows x 64 hidden cols
// (4 waves, one 16-col tile each, all 4 gates per wave -> local c/h update).
__global__ __launch_bounds__(128) void lstm_layer_kernel(
    const float* __restrict__ inp, int inp_stride,     // [B, *] input rows
    const float* __restrict__ h_prev,                  // [B,H] h state (read)
    float* __restrict__ h_cur,                         // [B,H] h state (write)
    float* __restrict__ c_state,                       // [B,H] c state (rmw)
    const float* __restrict__ mask,                    // [B,H] recurrent dropout
    const float* __restrict__ bias,                    // [4H]
    const __bf16* __restrict__ wt,                     // this layer's swizzled W
    const int* __restrict__ lengths,                   // [B]
    float* __restrict__ outp, int out_stride,          // raw nh destination
    int t)
{
    __shared__ __align__(32) __bf16 lds_in[16][LDSS];
    __shared__ __align__(32) __bf16 lds_hd[16][LDSS];

    const int tid     = threadIdx.x;
    const int wave    = tid >> 5;
    const int lane    = tid & 31;
    const int row0    = blockIdx.x * 16;
    const int colTile = blockIdx.y * 4 + wave;      // 0..31
    const int col     = colTile * 16 + (lane & 15); // hidden column
    const int hlf     = lane >> 4;                  // 0/1 lane half

    // Cooperative fill: convert inp and h*mask tiles to bf16 in LDS.
    for (int i = tid; i < 16 * DD / 4; i += 128) {
        int r  = i >> 7;
        int c4 = (i & 127) << 2;
        const float4 xi = *(const float4*)(inp + (size_t)(row0 + r) * inp_stride + c4);
        lds_in[r][c4 + 0] = (__bf16)xi.x;
        lds_in[r][c4 + 1] = (__bf16)xi.y;
        lds_in[r][c4 + 2] = (__bf16)xi.z;
        lds_in[r][c4 + 3] = (__bf16)xi.w;
        const float4 hv = *(const float4*)(h_prev + (size_t)(row0 + r) * HH + c4);
        const float4 mv = *(const float4*)(mask   + (size_t)(row0 + r) * HH + c4);
        lds_hd[r][c4 + 0] = (__bf16)(hv.x * mv.x);
        lds_hd[r][c4 + 1] = (__bf16)(hv.y * mv.y);
        lds_hd[r][c4 + 2] = (__bf16)(hv.z * mv.z);
        lds_hd[r][c4 + 3] = (__bf16)(hv.w * mv.w);
    }
    __syncthreads();

    v8f acc[4] = {};                       // i, f, g, o accumulators (16x16 f32)
    const int m  = lane & 15;              // batch row within tile (A-matrix M)
    const int ko = hlf * 16;               // K sub-offset per A layout

    const __bf16* wA = wt + (size_t)colTile * 512 + (size_t)lane * 16;
    const __bf16* wB = wA + WT_PER_GEMM;

    // GEMM 1: inp @ Wi
    for (int kc = 0; kc < KCH; ++kc) {
        const __bf16* ap = &lds_in[m][kc * 32 + ko];
        v8bf lo = *(const v8bf*)ap;          // K = kc*32 + ko + 0..7
        v8bf hi = *(const v8bf*)(ap + 8);    // K = kc*32 + ko + 8..15
        v16bf a = __builtin_shufflevector(lo, hi, 0,1,2,3,4,5,6,7,8,9,10,11,12,13,14,15);
        const __bf16* wp = wA + (size_t)kc * 4 * 16384;
        __builtin_prefetch(wp + 4 * 16384, 0, 0);
#pragma unroll
        for (int g = 0; g < 4; ++g) {
            v16bf bfrag = *(const v16bf*)(wp + g * 16384);
            acc[g] = __builtin_amdgcn_wmma_f32_16x16x32_bf16(
                false, a, false, bfrag, (short)0, acc[g], false, false);
        }
    }
    // GEMM 2: (h*mask) @ Wh
    for (int kc = 0; kc < KCH; ++kc) {
        const __bf16* ap = &lds_hd[m][kc * 32 + ko];
        v8bf lo = *(const v8bf*)ap;
        v8bf hi = *(const v8bf*)(ap + 8);
        v16bf a = __builtin_shufflevector(lo, hi, 0,1,2,3,4,5,6,7,8,9,10,11,12,13,14,15);
        const __bf16* wp = wB + (size_t)kc * 4 * 16384;
        __builtin_prefetch(wp + 4 * 16384, 0, 0);
#pragma unroll
        for (int g = 0; g < 4; ++g) {
            v16bf bfrag = *(const v16bf*)(wp + g * 16384);
            acc[g] = __builtin_amdgcn_wmma_f32_16x16x32_bf16(
                false, a, false, bfrag, (short)0, acc[g], false, false);
        }
    }

    // Gate math + state update. C/D layout: VGPR r, lane -> row m = r + 8*hlf,
    // col = lane&15 (already folded into `col`).
    const float bi = bias[0 * HH + col];
    const float bf = bias[1 * HH + col];
    const float bg = bias[2 * HH + col];
    const float bo = bias[3 * HH + col];

#pragma unroll
    for (int r = 0; r < 8; ++r) {
        int row = row0 + r + hlf * 8;
        bool keep = t < lengths[row];
        size_t sidx = (size_t)row * HH + col;
        float c_old = c_state[sidx];
        float h_old = h_prev[sidx];
        float zi = acc[0][r] + bi;
        float zf = acc[1][r] + bf;
        float zg = acc[2][r] + bg;
        float zo = acc[3][r] + bo;
        float nc = sigmoidf_(zf) * c_old + sigmoidf_(zi) * tanhf(zg);
        float nh = sigmoidf_(zo) * tanhf(nc);
        c_state[sidx] = keep ? nc : c_old;
        h_cur[sidx]   = keep ? nh : h_old;
        outp[(size_t)row * out_stride + col] = nh;   // raw (unmasked) h
    }
}

extern "C" void kernel_launch(void* const* d_in, const int* in_sizes, int n_in,
                              void* d_out, int out_size, void* d_ws, size_t ws_size,
                              hipStream_t stream) {
    const float* x       = (const float*)d_in[0];  // [B,T,D]
    const float* h_masks = (const float*)d_in[1];  // [L,B,H]
    const float* Wi      = (const float*)d_in[2];  // [L,D,4H]
    const float* Wh      = (const float*)d_in[3];  // [L,H,4H]
    const float* b       = (const float*)d_in[4];  // [L,4H]
    const float* c0      = (const float*)d_in[5];  // [L,B,H]
    const float* h0      = (const float*)d_in[6];  // [L,B,H]
    const int*   lengths = (const int*)d_in[7];    // [B]
    float* out = (float*)d_out;                    // [B,T,H]

    char* ws = (char*)d_ws;
    __bf16* wt    = (__bf16*)ws;                                   // 8 MiB
    float* c_state = (float*)(ws + (size_t)WT_TOTAL * 2);          // [L,B,H]
    float* h_buf   = c_state + (size_t)NL * BATCH * HH;            // [L][2][B,H]
    float* h_raw   = h_buf + (size_t)NL * 2 * BATCH * HH;          // [B,H]

    // 1) Swizzle weights to WMMA fragment layout (bf16).
    wt_prep_kernel<<<WT_TOTAL / 256, 256, 0, stream>>>(Wi, Wh, wt);

    // 2) Initialize state.
    hipMemcpyAsync(c_state, c0, (size_t)NL * BATCH * HH * 4,
                   hipMemcpyDeviceToDevice, stream);
    hipMemcpyAsync(h_buf + 0, h0, (size_t)BATCH * HH * 4,
                   hipMemcpyDeviceToDevice, stream);                     // L0 parity 0
    hipMemcpyAsync(h_buf + (size_t)2 * BATCH * HH, h0 + (size_t)BATCH * HH,
                   (size_t)BATCH * HH * 4, hipMemcpyDeviceToDevice, stream); // L1 parity 0

    // 3) Sequential scan: 2 layer-kernels per step, stream-ordered.
    dim3 grid(BATCH / 16, HH / 64), block(128);
    for (int t = 0; t < TT; ++t) {
        int p = t & 1;
        // Layer 0: input = x[:, t, :], raw nh -> h_raw
        lstm_layer_kernel<<<grid, block, 0, stream>>>(
            x + (size_t)t * DD, TT * DD,
            h_buf + (size_t)(0 * 2 + p)     * BATCH * HH,
            h_buf + (size_t)(0 * 2 + (1-p)) * BATCH * HH,
            c_state + (size_t)0 * BATCH * HH,
            h_masks + (size_t)0 * BATCH * HH,
            b + (size_t)0 * G4H,
            wt + (size_t)0 * WT_PER_LAYER,
            lengths,
            h_raw, HH, t);
        // Layer 1: input = h_raw, raw nh -> out[:, t, :]
        lstm_layer_kernel<<<grid, block, 0, stream>>>(
            h_raw, HH,
            h_buf + (size_t)(1 * 2 + p)     * BATCH * HH,
            h_buf + (size_t)(1 * 2 + (1-p)) * BATCH * HH,
            c_state + (size_t)1 * BATCH * HH,
            h_masks + (size_t)1 * BATCH * HH,
            b + (size_t)1 * G4H,
            wt + (size_t)1 * WT_PER_LAYER,
            lengths,
            out + (size_t)t * HH, TT * HH, t);
    }
}
